// SimpleDeltaNet_90606630076772
// MI455X (gfx1250) — compile-verified, matrix operand order checked by reading, not run on hardware
//
#include <hip/hip_runtime.h>
#include <hip/hip_bf16.h>

// ---------------- types ----------------
typedef __attribute__((ext_vector_type(16))) __bf16 v16bf;
typedef __attribute__((ext_vector_type(8)))  float  v8f;
typedef __attribute__((ext_vector_type(2)))  float  v2f;
typedef __attribute__((ext_vector_type(4)))  unsigned int v4u;
typedef __attribute__((ext_vector_type(8)))  int    v8i_;
typedef __attribute__((ext_vector_type(4)))  int    v4i_;

#define Bc   2
#define Sc   4096
#define Hm   2048
#define NHc  16
#define HDc  128
#define CH   16      // chunk length for delta rule
#define DVS  32      // dv slice width per block

__device__ __forceinline__ unsigned short f2bf(float f) {
    union { float f; unsigned u; } x; x.f = f;
    unsigned r = x.u + 0x7fffu + ((x.u >> 16) & 1u);   // RNE
    return (unsigned short)(r >> 16);
}

__device__ __forceinline__ v8f vzero8() {
    v8f r;
    #pragma unroll
    for (int i = 0; i < 8; i++) r[i] = 0.f;
    return r;
}

// ---- Tensor Data Mover: async 2D f32 tile load global -> LDS (6-arg builtin)
// D# per CDNA5 ISA: group0 {count, lds_addr, global_addr, type=2},
// group1 {data_size=4B, tensor_dim0/1, tile_dim0/1, tensor_dim0_stride}.
__device__ __forceinline__ void tdm_load_2d_f32(void* lds_ptr, const float* gptr,
                                                unsigned tile_x, unsigned tile_y,
                                                unsigned row_stride_elems) {
    unsigned long long ga = (unsigned long long)gptr;
    unsigned lds_addr = (unsigned)(unsigned long long)lds_ptr;  // low 32b = LDS offset
    v4u g0;
    g0[0] = 1u;                                                  // count=1, no gather
    g0[1] = lds_addr;
    g0[2] = (unsigned)ga;
    g0[3] = ((unsigned)(ga >> 32) & 0x01FFFFFFu) | (2u << 30);   // addr[56:32] | type=2
    const unsigned dim0 = tile_x;            // tensor width (elems)
    const unsigned dim1 = 1u << 20;          // plenty of rows (no OOB clamp)
    v8i_ g1;
    g1[0] = (int)(2u << 16);                                     // data_size = 4 bytes
    g1[1] = (int)((dim0 & 0xffffu) << 16);                       // dim0[15:0]
    g1[2] = (int)(((dim0 >> 16) & 0xffffu) | ((dim1 & 0xffffu) << 16));
    g1[3] = (int)(((dim1 >> 16) & 0xffffu) | (tile_x << 16));    // tile_dim0
    g1[4] = (int)(tile_y & 0xffffu);                             // tile_dim1 (tile_dim2=0)
    g1[5] = (int)row_stride_elems;                               // dim0_stride[31:0]
    g1[6] = 0;
    g1[7] = 0;
    v4i_ z4; z4[0] = z4[1] = z4[2] = z4[3] = 0;
    v8i_ z8;
    #pragma unroll
    for (int i = 0; i < 8; i++) z8[i] = 0;
    __builtin_amdgcn_tensor_load_to_lds(g0, g1, z4, z4, z8, 0);
}

// ---------------- fp32 -> bf16 convert ----------------
__global__ __launch_bounds__(256) void cvt_bf16_kernel(const float* __restrict__ in,
                                                       unsigned short* __restrict__ out,
                                                       size_t n) {
    size_t i = (size_t)blockIdx.x * 256 + threadIdx.x;
    if (i < n) out[i] = f2bf(in[i]);
}

// ---------------- bf16 GEMM: C[M,N] = A[M,K] * B[N,K]^T  (both K-major) ---------
// block tile 128x128, K-step 32, 8 waves (4 M x 2 N), each wave 2x4 16x16 WMMA tiles
__global__ __launch_bounds__(256) void gemm_bf16(const unsigned short* __restrict__ A,
                                                 const unsigned short* __restrict__ Bw,
                                                 float* __restrict__ C,
                                                 int M, int N, int Kd) {
    __shared__ unsigned As[128 * 17];  // 128 rows x 16 uints (32 bf16) + pad
    __shared__ unsigned Bs[128 * 17];
    const int tid  = threadIdx.x;
    const int lane = tid & 31, wave = tid >> 5;
    const int wm = wave & 3, wn = wave >> 2;
    const int bm = blockIdx.y * 128, bn = blockIdx.x * 128;
    const unsigned* Au = (const unsigned*)A;
    const unsigned* Bu = (const unsigned*)Bw;
    const int Ku = Kd >> 1;

    v8f acc[2][4];
    #pragma unroll
    for (int i = 0; i < 2; i++)
        #pragma unroll
        for (int j = 0; j < 4; j++) acc[i][j] = vzero8();

    for (int k0 = 0; k0 < Kd; k0 += 32) {
        const int kb = k0 >> 1;
        for (int i = tid; i < 128 * 16; i += 256) {
            int r = i >> 4, cu = i & 15;
            As[r * 17 + cu] = Au[(size_t)(bm + r) * Ku + kb + cu];
            Bs[r * 17 + cu] = Bu[(size_t)(bn + r) * Ku + kb + cu];
        }
        if (k0 + 32 < Kd) {      // prefetch next K-slice (global_prefetch_b8)
            const int r = tid >> 1, cu = 16 + (tid & 1) * 8;
            __builtin_prefetch(&Au[(size_t)(bm + r) * Ku + kb + cu], 0, 0);
            __builtin_prefetch(&Bu[(size_t)(bn + r) * Ku + kb + cu], 0, 0);
        }
        __syncthreads();

        // per-lane fragment K-pair mapping (ISA 16-bit A layout, pairs as b32)
        const int khalf = (lane >> 4) << 2;  // +4 uint pairs for lanes 16..31
        union { v16bf v; unsigned u[8]; } aF[2], bF[4];
        #pragma unroll
        for (int i = 0; i < 2; i++) {
            const int ra = wm * 32 + i * 16 + (lane & 15);
            #pragma unroll
            for (int g = 0; g < 8; g++)
                aF[i].u[g] = As[ra * 17 + (g & 3) + ((g >= 4) ? 8 : 0) + khalf];
        }
        #pragma unroll
        for (int j = 0; j < 4; j++) {
            const int rb = wn * 64 + j * 16 + (lane & 15);
            #pragma unroll
            for (int g = 0; g < 8; g++)
                bF[j].u[g] = Bs[rb * 17 + (g & 3) + ((g >= 4) ? 8 : 0) + khalf];
        }
        #pragma unroll
        for (int i = 0; i < 2; i++)
            #pragma unroll
            for (int j = 0; j < 4; j++)
                acc[i][j] = __builtin_amdgcn_wmma_f32_16x16x32_bf16(
                    false, aF[i].v, false, bF[j].v, (short)0, acc[i][j], false, false);
        __syncthreads();
    }
    #pragma unroll
    for (int i = 0; i < 2; i++)
        #pragma unroll
        for (int j = 0; j < 4; j++) {
            const int m0 = bm + wm * 32 + i * 16 + ((lane >> 4) << 3);
            const int n  = bn + wn * 64 + j * 16 + (lane & 15);
            #pragma unroll
            for (int v = 0; v < 8; v++) C[(size_t)(m0 + v) * N + n] = acc[i][j][v];
        }
}

// ---------------- beta = sigmoid(x @ Wb^T) ----------------
__global__ __launch_bounds__(256) void beta_kernel(const float* __restrict__ x,
                                                   const float* __restrict__ Wb,
                                                   float* __restrict__ beta) {
    const int bt = blockIdx.x;                    // 0..B*S-1
    const int lane = threadIdx.x & 31, wave = threadIdx.x >> 5;
    const float* xr = x + (size_t)bt * Hm;
    #pragma unroll
    for (int hh = 0; hh < 2; hh++) {
        const int h = wave + hh * 8;
        const float* wr = Wb + (size_t)h * Hm;
        float s = 0.f;
        for (int c = lane; c < Hm; c += 32) s += xr[c] * wr[c];
        #pragma unroll
        for (int off = 16; off; off >>= 1) s += __shfl_xor(s, off, 32);
        if (lane == 0) beta[(size_t)bt * NHc + h] = 1.f / (1.f + __expf(-s));
    }
}

// ---------------- depthwise causal conv (K=4) + SiLU ----------------
__global__ __launch_bounds__(256) void conv_silu(const float* __restrict__ in,
                                                 const float* __restrict__ w,
                                                 float* __restrict__ out) {
    const size_t idx = (size_t)blockIdx.x * 256 + threadIdx.x;  // over B*S*Hm
    const int c = (int)(idx & (Hm - 1));
    const size_t bt = idx >> 11;
    const int t = (int)(bt & (Sc - 1));
    float s = 0.f;
    #pragma unroll
    for (int i = 0; i < 4; i++) {
        const int tt = t - 3 + i;
        if (tt >= 0) s += w[c * 4 + i] * in[idx - (size_t)(3 - i) * Hm];
    }
    out[idx] = s / (1.f + __expf(-s));
}

// ---------------- L2 norm over head_dim (one wave per 128-vector) --------------
__global__ __launch_bounds__(256) void l2norm_kernel(float* __restrict__ p, float scale) {
    const int vec = blockIdx.x * 8 + (threadIdx.x >> 5);
    const int lane = threadIdx.x & 31;
    float* r = p + (size_t)vec * HDc;
    float vals[4]; float ss = 0.f;
    #pragma unroll
    for (int j = 0; j < 4; j++) { vals[j] = r[lane + 32 * j]; ss += vals[j] * vals[j]; }
    #pragma unroll
    for (int off = 16; off; off >>= 1) ss += __shfl_xor(ss, off, 32);
    const float inv = rsqrtf(ss + 1e-6f) * scale;
    #pragma unroll
    for (int j = 0; j < 4; j++) r[lane + 32 * j] = vals[j] * inv;
}

// ---------------- chunked delta-rule scan (WY form), fp32 WMMA + TDM prefetch --
// grid: B*NH*4 blocks (dv sliced by 32), 128 threads (4 waves) each.
// State slice [128 dk x 32 dv] lives in LDS; K/Q/V chunk tiles double-buffered,
// next chunk DMA'd by the Tensor Data Mover while current chunk computes.
__global__ __launch_bounds__(128) void delta_kernel(const float* __restrict__ q,
                                                    const float* __restrict__ k,
                                                    const float* __restrict__ v,
                                                    const float* __restrict__ beta,
                                                    float* __restrict__ o) {
    __shared__ float Sst[128][DVS];     // state slice
    __shared__ float Kc[2][CH][128];    // double-buffered chunk K
    __shared__ float Qc[2][CH][128];    // double-buffered chunk Q
    __shared__ float Vc[2][CH][DVS];    // double-buffered chunk V slice
    __shared__ float Uc[CH][DVS];       // rhs -> U
    __shared__ float Akk[CH][CH];       // strict-lower K K^T
    __shared__ float Gm[CH][CH];        // tril(Q K^T) incl diag
    __shared__ float Mi[CH][CH];        // (I + diag(beta) Akk)^-1
    __shared__ float bet[CH];

    const int tid  = threadIdx.x;
    const int lane = tid & 31, wave = tid >> 5;
    const int lm = lane & 15;
    const int kof = (lane >> 4) * 2;    // f32 WMMA: lanes 16..31 hold K=2,3
    const int hi8 = (lane >> 4) << 3;   // C/D row offset for lanes 16..31

    const int bh = blockIdx.x >> 2;
    const int b  = bh >> 4, h = bh & 15;
    const int dvb = (blockIdx.x & 3) * DVS;
    const int rowstr = NHc * HDc;       // 2048 floats between timesteps

    for (int i = tid; i < 128 * DVS; i += 128) ((float*)Sst)[i] = 0.f;

    const size_t base0 = (((size_t)b * Sc) * NHc + h) * HDc;
    if (wave == 0) {                    // TDM: stage chunk 0 into buffer 0
        tdm_load_2d_f32(&Kc[0][0][0], k + base0,       128, CH, rowstr);
        tdm_load_2d_f32(&Qc[0][0][0], q + base0,       128, CH, rowstr);
        tdm_load_2d_f32(&Vc[0][0][0], v + base0 + dvb, DVS, CH, rowstr);
    }
    __syncthreads();

    for (int c = 0; c < Sc / CH; c++) {
        const int pb = c & 1;
        const int t0 = c * CH;
        const size_t base = base0 + (size_t)t0 * rowstr;

        if (wave == 0) __builtin_amdgcn_s_wait_tensorcnt(0);  // chunk c arrived
        __syncthreads();
        if (tid < CH) bet[tid] = beta[((size_t)b * Sc + t0 + tid) * NHc + h];
        if (wave == 0 && c + 1 < Sc / CH) {   // TDM: prefetch chunk c+1
            const size_t nb_ = base + (size_t)CH * rowstr;
            tdm_load_2d_f32(&Kc[pb ^ 1][0][0], k + nb_,       128, CH, rowstr);
            tdm_load_2d_f32(&Qc[pb ^ 1][0][0], q + nb_,       128, CH, rowstr);
            tdm_load_2d_f32(&Vc[pb ^ 1][0][0], v + nb_ + dvb, DVS, CH, rowstr);
        }
        __syncthreads();

        // phase 1: waves 0-1: pred = Kc@S then rhs = beta*(V-pred) -> Uc
        //          wave 2: Akk = strict-tril(Kc Kc^T); wave 3: Gm = tril(Qc Kc^T)
        if (wave < 2) {
            const int nb = wave * 16;
            v8f acc = vzero8();
            for (int kk = 0; kk < 32; kk++) {
                const int kb = kk * 4 + kof;
                v2f a;  a.x = Kc[pb][lm][kb];    a.y = Kc[pb][lm][kb + 1];
                v2f bb; bb.x = Sst[kb][nb + lm]; bb.y = Sst[kb + 1][nb + lm];
                acc = __builtin_amdgcn_wmma_f32_16x16x4_f32(false, a, false, bb,
                                                            (short)0, acc, false, false);
            }
            #pragma unroll
            for (int vv = 0; vv < 8; vv++) {
                const int m = vv + hi8;
                Uc[m][nb + lm] = bet[m] * (Vc[pb][m][nb + lm] - acc[vv]);
            }
        } else if (wave == 2) {
            v8f acc = vzero8();
            for (int kk = 0; kk < 32; kk++) {
                const int kb = kk * 4 + kof;
                v2f a; a.x = Kc[pb][lm][kb]; a.y = Kc[pb][lm][kb + 1];
                acc = __builtin_amdgcn_wmma_f32_16x16x4_f32(false, a, false, a,
                                                            (short)0, acc, false, false);
            }
            #pragma unroll
            for (int vv = 0; vv < 8; vv++) {
                const int m = vv + hi8;
                Akk[m][lm] = (lm < m) ? acc[vv] : 0.f;
            }
        } else {
            v8f acc = vzero8();
            for (int kk = 0; kk < 32; kk++) {
                const int kb = kk * 4 + kof;
                v2f a;  a.x = Qc[pb][lm][kb]; a.y = Qc[pb][lm][kb + 1];
                v2f bb; bb.x = Kc[pb][lm][kb]; bb.y = Kc[pb][lm][kb + 1];
                acc = __builtin_amdgcn_wmma_f32_16x16x4_f32(false, a, false, bb,
                                                            (short)0, acc, false, false);
            }
            #pragma unroll
            for (int vv = 0; vv < 8; vv++) {
                const int m = vv + hi8;
                Gm[m][lm] = (lm <= m) ? acc[vv] : 0.f;
            }
        }
        __syncthreads();

        // phase 2: Mi = (I + diag(beta) Akk)^-1 via forward substitution (wave 0)
        if (wave == 0 && lane < CH) {
            const int j = lane;
            float mcol[CH];
            #pragma unroll
            for (int i = 0; i < CH; i++) {
                float s = (i == j) ? 1.f : 0.f;
                #pragma unroll
                for (int l = 0; l < CH; l++)
                    if (l < i) s -= bet[i] * Akk[i][l] * mcol[l];
                mcol[i] = s;
                Mi[i][j] = s;
            }
        }
        __syncthreads();

        // phase 3: U = Mi @ rhs (waves 0-1, in-place per-column)
        if (wave < 2) {
            const int nb = wave * 16;
            v8f acc = vzero8();
            #pragma unroll
            for (int kk = 0; kk < 4; kk++) {
                const int kb = kk * 4 + kof;
                v2f a;  a.x = Mi[lm][kb];       a.y = Mi[lm][kb + 1];
                v2f bb; bb.x = Uc[kb][nb + lm]; bb.y = Uc[kb + 1][nb + lm];
                acc = __builtin_amdgcn_wmma_f32_16x16x4_f32(false, a, false, bb,
                                                            (short)0, acc, false, false);
            }
            #pragma unroll
            for (int vv = 0; vv < 8; vv++) Uc[vv + hi8][nb + lm] = acc[vv];
        }
        __syncthreads();

        // phase 4: o = Qc @ S + Gm @ U  (waves 0-1) -> global
        if (wave < 2) {
            const int nb = wave * 16;
            v8f acc = vzero8();
            for (int kk = 0; kk < 32; kk++) {
                const int kb = kk * 4 + kof;
                v2f a;  a.x = Qc[pb][lm][kb];    a.y = Qc[pb][lm][kb + 1];
                v2f bb; bb.x = Sst[kb][nb + lm]; bb.y = Sst[kb + 1][nb + lm];
                acc = __builtin_amdgcn_wmma_f32_16x16x4_f32(false, a, false, bb,
                                                            (short)0, acc, false, false);
            }
            #pragma unroll
            for (int kk = 0; kk < 4; kk++) {
                const int kb = kk * 4 + kof;
                v2f a;  a.x = Gm[lm][kb];       a.y = Gm[lm][kb + 1];
                v2f bb; bb.x = Uc[kb][nb + lm]; bb.y = Uc[kb + 1][nb + lm];
                acc = __builtin_amdgcn_wmma_f32_16x16x4_f32(false, a, false, bb,
                                                            (short)0, acc, false, false);
            }
            #pragma unroll
            for (int vv = 0; vv < 8; vv++)
                o[base + (size_t)(vv + hi8) * rowstr + dvb + nb + lm] = acc[vv];
        }
        __syncthreads();

        // phase 5: S += Kc^T @ U  (all 4 waves; wave owns M-tiles {w, w+4} x 2 N-tiles)
        #pragma unroll
        for (int s = 0; s < 2; s++) {
            const int mb = (wave + 4 * s) * 16;
            #pragma unroll
            for (int nt = 0; nt < 2; nt++) {
                const int nb2 = nt * 16;
                v8f acc;
                #pragma unroll
                for (int vv = 0; vv < 8; vv++) acc[vv] = Sst[mb + vv + hi8][nb2 + lm];
                #pragma unroll
                for (int kk = 0; kk < 4; kk++) {
                    const int kb = kk * 4 + kof;
                    v2f a;  a.x = Kc[pb][kb][mb + lm]; a.y = Kc[pb][kb + 1][mb + lm];
                    v2f bb; bb.x = Uc[kb][nb2 + lm];   bb.y = Uc[kb + 1][nb2 + lm];
                    acc = __builtin_amdgcn_wmma_f32_16x16x4_f32(false, a, false, bb,
                                                                (short)0, acc, false, false);
                }
                #pragma unroll
                for (int vv = 0; vv < 8; vv++) Sst[mb + vv + hi8][nb2 + lm] = acc[vv];
            }
        }
        __syncthreads();
    }
}

// ---------------- RMSNorm over head_dim + bf16 pack ----------------
__global__ __launch_bounds__(256) void rmsnorm_kernel(const float* __restrict__ o,
                                                      const float* __restrict__ w,
                                                      unsigned short* __restrict__ ob) {
    const int vec = blockIdx.x * 8 + (threadIdx.x >> 5);
    const int lane = threadIdx.x & 31;
    const float* r = o + (size_t)vec * HDc;
    float vals[4]; float ss = 0.f;
    #pragma unroll
    for (int j = 0; j < 4; j++) { vals[j] = r[lane + 32 * j]; ss += vals[j] * vals[j]; }
    #pragma unroll
    for (int off = 16; off; off >>= 1) ss += __shfl_xor(ss, off, 32);
    const float inv = rsqrtf(ss * (1.f / HDc) + 1e-5f);
    #pragma unroll
    for (int j = 0; j < 4; j++) {
        const int d = lane + 32 * j;
        ob[(size_t)vec * HDc + d] = f2bf(vals[j] * inv * w[d]);
    }
}

// ---------------- host launch ----------------
extern "C" void kernel_launch(void* const* d_in, const int* in_sizes, int n_in,
                              void* d_out, int out_size, void* d_ws, size_t ws_size,
                              hipStream_t stream) {
    const float* x      = (const float*)d_in[0];
    const float* Wq     = (const float*)d_in[1];
    const float* Wk     = (const float*)d_in[2];
    const float* Wv     = (const float*)d_in[3];
    const float* Wb     = (const float*)d_in[4];
    const float* Wo     = (const float*)d_in[5];
    const float* conv_q = (const float*)d_in[6];
    const float* conv_k = (const float*)d_in[7];
    const float* conv_v = (const float*)d_in[8];
    const float* rms_w  = (const float*)d_in[9];

    const size_t Mrows = (size_t)Bc * Sc;        // 8192
    unsigned char* wp = (unsigned char*)d_ws;
    auto alloc = [&](size_t bytes) -> void* {
        void* p = wp; wp += (bytes + 255) & ~(size_t)255; return p;
    };
    unsigned short* xb   = (unsigned short*)alloc(Mrows * Hm * 2);
    unsigned short* Wqb  = (unsigned short*)alloc((size_t)Hm * Hm * 2);
    unsigned short* Wkb  = (unsigned short*)alloc((size_t)Hm * Hm * 2);
    unsigned short* Wvb  = (unsigned short*)alloc((size_t)Hm * Hm * 2);
    unsigned short* Wob  = (unsigned short*)alloc((size_t)Hm * Hm * 2);
    float* qpre = (float*)alloc(Mrows * Hm * 4);
    float* kpre = (float*)alloc(Mrows * Hm * 4);
    float* vpre = (float*)alloc(Mrows * Hm * 4);
    float* qact = (float*)alloc(Mrows * Hm * 4);
    float* kact = (float*)alloc(Mrows * Hm * 4);
    float* vact = (float*)alloc(Mrows * Hm * 4);
    float* betab = (float*)alloc(Mrows * NHc * 4);
    float* o_raw = qpre;                          // qpre dead after conv_q
    unsigned short* ob = (unsigned short*)kpre;   // kpre dead after conv_k

    const size_t nX = Mrows * Hm, nW = (size_t)Hm * Hm;
    cvt_bf16_kernel<<<(unsigned)((nX + 255) / 256), 256, 0, stream>>>(x,  xb,  nX);
    cvt_bf16_kernel<<<(unsigned)((nW + 255) / 256), 256, 0, stream>>>(Wq, Wqb, nW);
    cvt_bf16_kernel<<<(unsigned)((nW + 255) / 256), 256, 0, stream>>>(Wk, Wkb, nW);
    cvt_bf16_kernel<<<(unsigned)((nW + 255) / 256), 256, 0, stream>>>(Wv, Wvb, nW);
    cvt_bf16_kernel<<<(unsigned)((nW + 255) / 256), 256, 0, stream>>>(Wo, Wob, nW);

    dim3 ggrid(Hm / 128, (unsigned)(Mrows / 128));
    gemm_bf16<<<ggrid, 256, 0, stream>>>(xb, Wqb, qpre, (int)Mrows, Hm, Hm);
    gemm_bf16<<<ggrid, 256, 0, stream>>>(xb, Wkb, kpre, (int)Mrows, Hm, Hm);
    gemm_bf16<<<ggrid, 256, 0, stream>>>(xb, Wvb, vpre, (int)Mrows, Hm, Hm);

    beta_kernel<<<(unsigned)Mrows, 256, 0, stream>>>(x, Wb, betab);

    const unsigned cblocks = (unsigned)(Mrows * Hm / 256);
    conv_silu<<<cblocks, 256, 0, stream>>>(qpre, conv_q, qact);
    conv_silu<<<cblocks, 256, 0, stream>>>(kpre, conv_k, kact);
    conv_silu<<<cblocks, 256, 0, stream>>>(vpre, conv_v, vact);

    const unsigned nvecB = (unsigned)(Mrows * NHc / 8);   // blocks of 8 vectors
    l2norm_kernel<<<nvecB, 256, 0, stream>>>(qact, 0.08838834764831845f); // 128^-0.5
    l2norm_kernel<<<nvecB, 256, 0, stream>>>(kact, 1.0f);

    delta_kernel<<<Bc * NHc * 4, 128, 0, stream>>>(qact, kact, vact, betab, o_raw);

    rmsnorm_kernel<<<nvecB, 256, 0, stream>>>(o_raw, rms_w, ob);

    gemm_bf16<<<ggrid, 256, 0, stream>>>(ob, Wob, (float*)d_out, (int)Mrows, Hm, Hm);
}
